// WindowAttention_73693048865419
// MI455X (gfx1250) — matrix operand order repaired and checked by reading, not compiled, not run on hardware
//
#include <hip/hip_runtime.h>

typedef __attribute__((ext_vector_type(16))) _Float16 v16h;
typedef __attribute__((ext_vector_type(8)))  _Float16 v8h;
typedef __attribute__((ext_vector_type(8)))  float    v8f;

// workspace layout (bytes)
#define WS_WT   0         // f16 WT[576][192]   (qkv weights, transposed)  221184 B
#define WS_PT   221184    // f16 projT[192][192]                            73728 B
#define WS_BC   294912    // f32 biasC[576]                                  2304 B
#define WS_BM   297216    // f32 biasFrag[6][4][4][32][8] (C-frag order)    98304 B
// total 395520 B

static __device__ __forceinline__ v8f wmma16(v16h a, v16h b, v8f c) {
  return __builtin_amdgcn_wmma_f32_16x16x32_f16(false, a, false, b, (short)0, c,
                                                false, false);
}

// A fragment: lane holds row M=L%16, K = (L/16)*8 + {0..7} then +16..+23
static __device__ __forceinline__ v16h frag_a(const _Float16* p) {
  v8h lo = *(const v8h*)p;
  v8h hi = *(const v8h*)(p + 16);
  return __builtin_shufflevector(lo, hi, 0,1,2,3,4,5,6,7,8,9,10,11,12,13,14,15);
}
// B fragment: lane holds col N=L%16, K = (L/16)*8 + {0..7} then +16..+23 (N-major storage)
static __device__ __forceinline__ v16h frag_b(const _Float16* p) {
  v8h lo = *(const v8h*)p;
  v8h hi = *(const v8h*)(p + 16);
  return __builtin_shufflevector(lo, hi, 0,1,2,3,4,5,6,7,8,9,10,11,12,13,14,15);
}

__global__ __launch_bounds__(256) void win_attn_prep(
    const float* __restrict__ wq, const float* __restrict__ bq,
    const float* __restrict__ wkv, const float* __restrict__ bkv,
    const float* __restrict__ bias_table, const float* __restrict__ proj_w,
    unsigned char* __restrict__ ws)
{
  _Float16* WT = (_Float16*)(ws + WS_WT);
  _Float16* PT = (_Float16*)(ws + WS_PT);
  float*    BC = (float*)(ws + WS_BC);
  float*    BM = (float*)(ws + WS_BM);
  int i = blockIdx.x * 256 + threadIdx.x;
  if (i < 110592) {                       // WT[n][k] = W[k][n], n over [q|k|v]
    int n = i / 192, k = i - n * 192;
    float w = (n < 192) ? wq[k * 192 + n] : wkv[k * 384 + (n - 192)];
    WT[i] = (_Float16)w;
  } else if (i < 147456) {                // projT[n][k] = proj_w[k][n]
    int j = i - 110592;
    int n = j / 192, k = j - n * 192;
    PT[j] = (_Float16)proj_w[k * 192 + n];
  } else if (i < 148032) {                // combined bias vector
    int n = i - 147456;
    BC[n] = (n < 192) ? bq[n] : bkv[n - 192];
  } else if (i < 172608) {                // bias in WMMA C-fragment order:
    int o = i - 148032;                   // [h][mt][jt][lane][r]
    int r    = o & 7;
    int lane = (o >> 3) & 31;
    int jt   = (o >> 8) & 3;
    int mt   = (o >> 10) & 3;
    int h    = o >> 12;
    int p = mt * 16 + r + (lane >> 4) * 8;    // query index
    int q = jt * 16 + (lane & 15);            // key index
    int idx = ((p & 7) - (q & 7) + 7) * 15 + ((p >> 3) - (q >> 3) + 7);
    BM[o] = bias_table[idx * 6 + h];
  }
}

__global__ __launch_bounds__(256) void win_attn_main(
    const float* __restrict__ x, const float* __restrict__ proj_b,
    const unsigned char* __restrict__ ws, float* __restrict__ out)
{
  constexpr int XS = 200;   // padded row stride (halfs) for [64][192] buffers
  constexpr int VTS = 72;   // padded token stride for vT[6][32][64]
  constexpr int STGS = 72;  // padded stride for per-wave P staging [16][64]

  __shared__ _Float16 sm[61440];          // 122880 B (<320 KB/WGP)
  _Float16* xo_sh  = sm;                  // x (stage1 A), reused as attn-out
  _Float16* q_sh   = sm + 12800;          // [64][200]
  _Float16* k_sh   = sm + 25600;          // [64][200]
  _Float16* vt_sh  = sm + 38400;          // [6][32][72]
  _Float16* stg_sh = sm + 52224;          // [8][16][72]

  const _Float16* WT = (const _Float16*)(ws + WS_WT);
  const _Float16* PT = (const _Float16*)(ws + WS_PT);
  const float*    BC = (const float*)(ws + WS_BC);
  const float*    BM = (const float*)(ws + WS_BM);

  const int tid = threadIdx.x;
  const int lane = tid & 31;
  // wave id is uniform: pin it (and everything derived) into SGPRs
  const int wvS = __builtin_amdgcn_readfirstlane(tid >> 5);
  const int mt  = wvS & 3;                // fixed M-tile per wave
  const int lmod = lane & 15, lhalf = lane >> 4;
  const int kOffA = lhalf * 8;            // A-frag K offset for this lane
  const int kOffB = lhalf * 8;            // B-frag K offset for this lane
  const int blk = blockIdx.x;

  // ---- load x tile [64][192] fp32 -> f16 LDS (row-major, padded) ----
  const float* xg = x + (size_t)blk * 12288;
  #pragma unroll
  for (int i = 0; i < 12; ++i) {
    int v = tid + i * 256;                // 3072 float4 per block
    float4 f = ((const float4*)xg)[v];
    int fi = v * 4, m = fi / 192, c = fi - m * 192;
    _Float16* d = &xo_sh[m * XS + c];
    d[0] = (_Float16)f.x; d[1] = (_Float16)f.y;
    d[2] = (_Float16)f.z; d[3] = (_Float16)f.w;
  }
  __syncthreads();

  // ---- Stage 1: [64x192] @ [192x576] -> q,k,vT with bias+leaky(+scale) ----
  {
    const _Float16* arow = &xo_sh[(mt * 16 + lmod) * XS];
    v16h afr[6];                          // A fragments: loop-invariant per wave
    #pragma unroll
    for (int kc = 0; kc < 6; ++kc) afr[kc] = frag_a(arow + kc * 32 + kOffA);

    for (int nt = wvS >> 2; nt < 36; nt += 2) {   // scalar loop control
      const int n = nt * 16 + lmod;       // output channel for this lane
      v8f acc = {};
      const _Float16* wrow = WT + n * 192;
      #pragma unroll
      for (int kc = 0; kc < 6; ++kc)
        acc = wmma16(afr[kc], frag_b(wrow + kc * 32 + kOffB), acc);

      const float bias = BC[n];
      float res[8];
      #pragma unroll
      for (int r = 0; r < 8; ++r) {
        float v = acc[r] + bias;
        res[r] = v >= 0.f ? v : 0.2f * v; // LeakyReLU(0.2)
      }
      // nt is an SGPR -> pure scalar branches
      if (nt < 12) {
        #pragma unroll
        for (int r = 0; r < 8; ++r)
          q_sh[(mt * 16 + r + lhalf * 8) * XS + n] =
              (_Float16)(res[r] * 0.17677669529663687f);
      } else if (nt < 24) {
        #pragma unroll
        for (int r = 0; r < 8; ++r)
          k_sh[(mt * 16 + r + lhalf * 8) * XS + (n - 192)] = (_Float16)res[r];
      } else {
        const int c = n - 384;            // v: store transposed [h][d][tok]
        _Float16* vrow = &vt_sh[(c >> 5) * (32 * VTS) + (c & 31) * VTS + mt * 16];
        #pragma unroll
        for (int r = 0; r < 8; ++r)
          vrow[r + lhalf * 8] = (_Float16)res[r];
      }
    }
  }
  __syncthreads();

  // ---- Stage 2: per (head, 16-row block): scores, softmax, P @ V ----
  _Float16* stg = stg_sh + wvS * 16 * STGS;
  for (int h = wvS >> 2; h < 6; h += 2) {  // scalar loop; mt fixed per wave
    v16h aq = frag_a(&q_sh[(mt * 16 + lmod) * XS + h * 32 + kOffA]);
    v8f S[4];
    const float* bmf = BM + (h * 4 + mt) * 1024 + lane * 8;  // C-frag order
    #pragma unroll
    for (int jt = 0; jt < 4; ++jt) {      // D=32 == one K-chunk
      v16h bk = frag_b(&k_sh[(jt * 16 + lmod) * XS + h * 32 + kOffB]);
      v8f z = {};
      S[jt] = wmma16(aq, bk, z);
      v8f bv = *(const v8f*)(bmf + jt * 256);   // two b128 loads, coalesced
      S[jt] = S[jt] + bv;
    }

    // softmax over 64 keys: row = (r, lane-half); reduce across 16 lanes + 4 tiles
    #pragma unroll
    for (int r = 0; r < 8; ++r) {
      float mx = fmaxf(fmaxf(S[0][r], S[1][r]), fmaxf(S[2][r], S[3][r]));
      for (int m = 1; m < 16; m <<= 1) mx = fmaxf(mx, __shfl_xor(mx, m, 32));
      float sum = 0.f;
      #pragma unroll
      for (int jt = 0; jt < 4; ++jt) {
        float e = __expf(S[jt][r] - mx);
        S[jt][r] = e; sum += e;
      }
      for (int m = 1; m < 16; m <<= 1) sum += __shfl_xor(sum, m, 32);
      const float inv = 1.f / sum;
      #pragma unroll
      for (int jt = 0; jt < 4; ++jt)       // stage P as f16 A-operand rows
        stg[(r + lhalf * 8) * STGS + jt * 16 + lmod] = (_Float16)(S[jt][r] * inv);
    }

    // P[16x64] @ V[64x32]
    v8f O0 = {}, O1 = {};
    #pragma unroll
    for (int kc = 0; kc < 2; ++kc) {
      v16h ap = frag_a(&stg[lmod * STGS + kc * 32 + kOffA]);
      v16h b0 = frag_b(&vt_sh[h * (32 * VTS) + lmod * VTS + kc * 32 + kOffB]);
      v16h b1 = frag_b(&vt_sh[h * (32 * VTS) + (16 + lmod) * VTS + kc * 32 + kOffB]);
      O0 = wmma16(ap, b0, O0);
      O1 = wmma16(ap, b1, O1);
    }
    #pragma unroll
    for (int r = 0; r < 8; ++r) {
      const int row = mt * 16 + r + lhalf * 8;
      xo_sh[row * XS + h * 32 + lmod]      = (_Float16)O0[r];
      xo_sh[row * XS + h * 32 + 16 + lmod] = (_Float16)O1[r];
    }
  }
  __syncthreads();

  // ---- Stage 3: [64x192] @ projT -> fp32 out ----
  {
    float* og = out + (size_t)blk * 12288;
    const _Float16* arow = &xo_sh[(mt * 16 + lmod) * XS];
    v16h afr[6];                          // attention output rows, invariant
    #pragma unroll
    for (int kc = 0; kc < 6; ++kc) afr[kc] = frag_a(arow + kc * 32 + kOffA);

    for (int nt = wvS >> 2; nt < 12; nt += 2) {
      const int n = nt * 16 + lmod;
      v8f acc = {};
      const _Float16* prow = PT + n * 192;
      #pragma unroll
      for (int kc = 0; kc < 6; ++kc)
        acc = wmma16(afr[kc], frag_b(prow + kc * 32 + kOffB), acc);
      const float pb = proj_b[n];
      #pragma unroll
      for (int r = 0; r < 8; ++r)
        og[(mt * 16 + r + lhalf * 8) * 192 + n] = acc[r] + pb;
    }
  }
}

extern "C" void kernel_launch(void* const* d_in, const int* in_sizes, int n_in,
                              void* d_out, int out_size, void* d_ws, size_t ws_size,
                              hipStream_t stream) {
  (void)in_sizes; (void)n_in; (void)out_size; (void)ws_size;
  const float* x   = (const float*)d_in[0];
  const float* wq  = (const float*)d_in[1];
  const float* bq  = (const float*)d_in[2];
  const float* wkv = (const float*)d_in[3];
  const float* bkv = (const float*)d_in[4];
  const float* bt  = (const float*)d_in[5];
  const float* pw  = (const float*)d_in[6];
  const float* pb  = (const float*)d_in[7];
  unsigned char* ws = (unsigned char*)d_ws;

  // prepass: 172608 work items (weight transpose/convert + bias fragments)
  win_attn_prep<<<675, 256, 0, stream>>>(wq, bq, wkv, bkv, bt, pw, ws);
  // fused attention: one workgroup per window
  win_attn_main<<<4096, 256, 0, stream>>>(x, pb, ws, (float*)d_out);
}
